// DeepCausalQualityVGG_21766894256609
// MI455X (gfx1250) — matrix operand order repaired and checked by reading, not compiled, not run on hardware
//
#include <hip/hip_runtime.h>
#include <hip/hip_bf16.h>
#include <stdint.h>

typedef __attribute__((ext_vector_type(16))) _Float16 v16h;
typedef __attribute__((ext_vector_type(8)))  _Float16 v8h;
typedef __attribute__((ext_vector_type(8)))  float    v8f;
typedef __attribute__((ext_vector_type(4)))  float    v4f;
typedef __attribute__((ext_vector_type(4)))  uint32_t u32x4;
typedef __attribute__((ext_vector_type(8)))  int      i32x8;
typedef __attribute__((ext_vector_type(4)))  int      i32x4;

// Tensor Data Mover availability (device pass only; host pass sees fallback)
#if defined(__has_builtin)
#  if __has_builtin(__builtin_amdgcn_tensor_load_to_lds) && __has_builtin(__builtin_amdgcn_s_wait_tensorcnt)
#    define USE_TDM 1
#  endif
#endif
#ifndef USE_TDM
#  define USE_TDM 0
#endif

// ---------------------------------------------------------------------------
// Utility kernels
// ---------------------------------------------------------------------------
__global__ void zeroK(float* p, int n) {
    int i = blockIdx.x * blockDim.x + threadIdx.x;
    if (i < n) p[i] = 0.0f;
}

// NCHW f32 -> NHWC f32 (for stage-0 distance on raw inputs)
__global__ void nchwToNhwcF32(const float* __restrict__ in, float* __restrict__ out,
                              int C, int H, int W) {
    int i = blockIdx.x * blockDim.x + threadIdx.x;
    if (i >= C * H * W) return;
    const int c   = i % C;
    const int pix = i / C;
    const int py  = pix / W;
    const int px  = pix - py * W;
    out[i] = in[((size_t)c * H + py) * W + px];
}

// NCHW f32 (C=3) -> NHWC f16 padded to 32 channels (conv layer-0 input)
__global__ void padInputNHWC16(const float* __restrict__ in, _Float16* __restrict__ out,
                               int H, int W) {
    int i = blockIdx.x * blockDim.x + threadIdx.x;
    if (i >= H * W * 32) return;
    const int c   = i & 31;
    const int pix = i >> 5;
    const int py  = pix / W;
    const int px  = pix - py * W;
    _Float16 v = (_Float16)0.0f;
    if (c < 3) v = (_Float16)in[((size_t)c * H + py) * W + px];
    out[i] = v;
}

// ---------------------------------------------------------------------------
// Weight repack: f32 (Cout,Cin,3,3) -> f16 WMMA A-fragment order, chunked as
// chunk ch = pos*KcC + cc  (pos = fy*3+fx, cc = cin/32).  Within a 16x32 tile
// (512 halves): lane l, half j ->  m=(l&15),
//   cin = cc*32 + (l>=16 ? 16 : 0) + j      (contiguous K, matches B)
// ---------------------------------------------------------------------------
__global__ void repackW(const float* __restrict__ w, _Float16* __restrict__ pk,
                        int Cout, int Cin, int KcC, int total) {
    int idx = blockIdx.x * blockDim.x + threadIdx.x;
    if (idx >= total) return;
    const int within = idx & 511;
    const int tile   = idx >> 9;
    const int lane   = within >> 4;
    const int j      = within & 15;
    const int CPR = 9 * KcC;                 // chunks per tile-row
    const int mt  = tile / CPR;
    const int ch  = tile - mt * CPR;
    const int pos = ch / KcC;
    const int cc  = ch - pos * KcC;
    const int m   = mt * 16 + (lane & 15);
    const int cin = cc * 32 + ((lane >> 4) << 4) + j;
    _Float16 v = (_Float16)0.0f;
    if (cin < Cin) v = (_Float16)w[((size_t)m * Cin + cin) * 9 + pos];
    pk[idx] = v;
}

// ---------------------------------------------------------------------------
// A-chunk staging into LDS: 4 tiles x 512 halves (4 KB), tiles strided by
// CPR*512 elements in global.  TDM path: one 2-D tensor_load_to_lds
// (tile 512x4, data_size=2B).  Fallback: cooperative v16h copies.
// ---------------------------------------------------------------------------
#if USE_TDM
__device__ inline void fillA_tdm(_Float16* dst, const _Float16* wpk,
                                 int mt0, int CPR, int ch) {
    const uint64_t ga = (uint64_t)(uintptr_t)(wpk + (((size_t)mt0 * CPR + ch) << 9));
    const uint32_t ldsOff = (uint32_t)(uintptr_t)dst;
    // D# group0: count=1 | lds_addr | global_addr | type=2
    u32x4 g0 = { 1u, ldsOff, (uint32_t)ga,
                 (uint32_t)((ga >> 32) & 0x01FFFFFFu) | (2u << 30) };
    // D# group1: data_size=2B; tensor_dim0=512, tensor_dim1=4;
    //            tile_dim0=512, tile_dim1=4; tensor_dim0_stride=CPR*512
    i32x8 g1 = { (int)0x00010000u, (int)(512u << 16), (int)(4u << 16),
                 (int)(512u << 16), 4, (int)(CPR << 9), 0, 0 };
    i32x4 z4 = { 0, 0, 0, 0 };
#if __clang_major__ >= 23
    i32x8 z8 = { 0, 0, 0, 0, 0, 0, 0, 0 };
    __builtin_amdgcn_tensor_load_to_lds(g0, g1, z4, z4, z8, 0);
#else
    __builtin_amdgcn_tensor_load_to_lds(g0, g1, z4, z4, 0);
#endif
}
#endif

__device__ inline void fillA(_Float16* dst, const _Float16* wpk,
                             int mt0, int CPR, int ch, int tid) {
#if USE_TDM
    if ((tid >> 5) == 0) fillA_tdm(dst, wpk, mt0, CPR, ch);
#else
    const _Float16* src = wpk + (((size_t)(mt0 + (tid >> 5)) * CPR + ch) << 9)
                              + ((size_t)(tid & 31) << 4);
    *reinterpret_cast<v16h*>(dst + ((size_t)tid << 4)) =
        *reinterpret_cast<const v16h*>(src);
#endif
}

__device__ inline void waitA(int tid) {
#if USE_TDM
    if ((tid >> 5) == 0) __builtin_amdgcn_s_wait_tensorcnt((short)0);
#endif
}

// ---------------------------------------------------------------------------
// Implicit-GEMM 3x3 conv (stride 1, pad 1) + bias + ReLU via WMMA f16->f32.
// NHWC activations (channels padded to mult of 32).  128-thread block =
// 4 waves = 64(Cout) x 64(pixel) tile; A fragments double-buffered in LDS
// (TDM-filled), shared by all waves; each wave streams its own B.
// A fragments for all 4 M-tiles are loaded before the 4 WMMAs so the
// ds_loads batch and overlap the matrix pipe; NHWC makes the epilogue
// vector stores (2x b128 f32 + 1x b128 f16 per tile).
// ---------------------------------------------------------------------------
__global__ void __launch_bounds__(128)
convWMMA(const _Float16* __restrict__ in16, const _Float16* __restrict__ wpk,
         const float* __restrict__ bias, float* __restrict__ out32,
         _Float16* __restrict__ out16, int H, int W, int Cinp, int KcC, int Cout) {
    __shared__ __align__(32) _Float16 abuf[2][2048];   // 2 x 4KB A chunks
    const int tid  = threadIdx.x;
    const int lane = tid & 31;
    const int wv   = tid >> 5;
    const int n0   = blockIdx.x * 64 + wv * 16;
    const int m0   = blockIdx.y * 64;
    const int mt0  = m0 >> 4;
    const int CPR  = 9 * KcC;

    const int n  = n0 + (lane & 15);       // pixel (always < H*W)
    const int py = n / W;
    const int px = n - py * W;
    const int kHi = ((lane >> 4) << 4);    // B layout: lanes>=16 hold K 16..31

    fillA(&abuf[0][0], wpk, mt0, CPR, 0, tid);   // prologue: chunk 0

    v8f acc[4] = {{}, {}, {}, {}};
    int ch = 0;

    #pragma unroll
    for (int fy = 0; fy < 3; ++fy) {
        const int iy = py + fy - 1;
        #pragma unroll
        for (int fx = 0; fx < 3; ++fx) {
            const int ix    = px + fx - 1;
            const bool valid = (iy >= 0) & (iy < H) & (ix >= 0) & (ix < W);
            const int iyc = iy < 0 ? 0 : (iy >= H ? H - 1 : iy);
            const int ixc = ix < 0 ? 0 : (ix >= W ? W - 1 : ix);
            const _Float16* bp = in16 + (size_t)(iyc * W + ixc) * Cinp + kHi;
            for (int cc = 0; cc < KcC; ++cc, ++ch) {
                const int cur = ch & 1;
                waitA(tid);                         // TDM fill of abuf[cur] done
                __syncthreads();                    // publish; prior reads done
                if (ch + 1 < CPR)
                    fillA(&abuf[(ch + 1) & 1][0], wpk, mt0, CPR, ch + 1, tid);

                v16h b = *reinterpret_cast<const v16h*>(bp + cc * 32);
                if (!valid) b = (v16h){};           // zero-pad tap

                v16h af[4];
                #pragma unroll
                for (int t = 0; t < 4; ++t)         // batch all 8 ds_load_b128
                    af[t] = *reinterpret_cast<const v16h*>(
                        &abuf[cur][(t << 9) + (lane << 4)]);
                #pragma unroll
                for (int t = 0; t < 4; ++t)
                    acc[t] = __builtin_amdgcn_wmma_f32_16x16x32_f16(
                        false, af[t], false, b, (short)0, acc[t], false, false);
            }
        }
    }

    // C/D layout: VGPR r -> M = r + 8*(lane>=16), N = lane&15.  NHWC store:
    // each tile's 8 M-values are contiguous -> vector stores.
    const size_t outRow = (size_t)n * Cout;
    #pragma unroll
    for (int t = 0; t < 4; ++t) {
        const int mBase = m0 + t * 16 + ((lane >> 4) << 3);
        float f[8];
        #pragma unroll
        for (int r = 0; r < 8; ++r) {
            float v = acc[t][r] + bias[mBase + r];
            f[r] = v > 0.0f ? v : 0.0f;
        }
        v4f lo = { f[0], f[1], f[2], f[3] };
        v4f hi = { f[4], f[5], f[6], f[7] };
        *reinterpret_cast<v4f*>(&out32[outRow + mBase])     = lo;
        *reinterpret_cast<v4f*>(&out32[outRow + mBase + 4]) = hi;
        v8h h8 = { (_Float16)f[0], (_Float16)f[1], (_Float16)f[2], (_Float16)f[3],
                   (_Float16)f[4], (_Float16)f[5], (_Float16)f[6], (_Float16)f[7] };
        *reinterpret_cast<v8h*>(&out16[outRow + mBase]) = h8;
    }
}

// ---------------------------------------------------------------------------
// Hanning L2 pool (NHWC): depthwise 3x3 normalized hanning on x^2, stride 2,
// pad 1, sqrt(+1e-12).  Writes f16 NHWC (feed for next conv stage).
// ---------------------------------------------------------------------------
__global__ void l2poolK(const float* __restrict__ in, _Float16* __restrict__ out16,
                        int C, int H, int W) {
    const int H2 = H >> 1, W2 = W >> 1;
    const int total = C * H2 * W2;
    int idx = blockIdx.x * blockDim.x + threadIdx.x;
    if (idx >= total) return;
    const int c   = idx % C;
    const int pix = idx / C;
    const int oy  = pix / W2;
    const int ox  = pix - oy * W2;
    const float g[3] = {0.25f, 0.5f, 0.25f};
    float acc = 0.0f;
    #pragma unroll
    for (int fy = 0; fy < 3; ++fy) {
        const int iy = 2 * oy + fy - 1;
        if (iy < 0 || iy >= H) continue;
        #pragma unroll
        for (int fx = 0; fx < 3; ++fx) {
            const int ix = 2 * ox + fx - 1;
            if (ix < 0 || ix >= W) continue;
            const float v = in[(size_t)(iy * W + ix) * C + c];
            acc += (g[fy] * g[fx] * 0.25f) * v * v;
        }
    }
    out16[idx] = (_Float16)sqrtf(acc + 1e-12f);
}

// ---------------------------------------------------------------------------
// Distance scratch (floats at start of d_ws):
//   scal[4]: 0=sum(X) 1=sum(Y) 2=sum((xpdf+ypdf)/2) 3=sum((xpdf-ypdf)/2)
//   otd[10*512], l1[512]
// ---------------------------------------------------------------------------
__global__ void reduceRaw(const float* __restrict__ X, const float* __restrict__ Y,
                          int n, float* scal) {
    __shared__ float sx[256], sy[256];
    const int t = threadIdx.x;
    float ax = 0.0f, ay = 0.0f;
    for (int i = blockIdx.x * 256 + t; i < n; i += gridDim.x * 256) {
        ax += X[i]; ay += Y[i];
    }
    sx[t] = ax; sy[t] = ay; __syncthreads();
    for (int st = 128; st > 0; st >>= 1) {
        if (t < st) { sx[t] += sx[t + st]; sy[t] += sy[t + st]; }
        __syncthreads();
    }
    if (t == 0) { atomicAdd(&scal[0], sx[0]); atomicAdd(&scal[1], sy[0]); }
}

__global__ void reducePdf(const float* __restrict__ X, const float* __restrict__ Y,
                          int n, float* scal) {
    __shared__ float sa[256], sb[256];
    const int t = threadIdx.x;
    const float s0 = scal[0], s1 = scal[1];
    float aa = 0.0f, ab = 0.0f;
    for (int i = blockIdx.x * 256 + t; i < n; i += gridDim.x * 256) {
        const float xv = X[i], yv = Y[i];
        const float xp = xv * xv / (s0 + 1e-6f);
        const float yp = yv * yv / (s1 + 1e-6f);
        aa += 0.5f * (xp + yp);
        ab += 0.5f * (xp - yp);
    }
    sa[t] = aa; sb[t] = ab; __syncthreads();
    for (int st = 128; st > 0; st >>= 1) {
        if (t < st) { sa[t] += sa[t + st]; sb[t] += sb[t + st]; }
        __syncthreads();
    }
    if (t == 0) { atomicAdd(&scal[2], sa[0]); atomicAdd(&scal[3], sb[0]); }
}

__device__ inline unsigned mixu(unsigned x) {
    x ^= x >> 16; x *= 0x7feb352dU; x ^= x >> 15; x *= 0x846ca68bU; x ^= x >> 16;
    return x;
}
__device__ inline float gaussn(int stage, int s, int c, int m, int j) {
    unsigned h = mixu(0x9E3779B9U * (unsigned)stage ^ 0x85EBCA6BU * (unsigned)s ^
                      0xC2B2AE35U * (unsigned)c ^ 0x27D4EB2FU * (unsigned)m ^
                      0x165667B1U * (unsigned)j ^ 0xA511E9B3U);
    unsigned h2 = mixu(h ^ 0x68E31DA4U);
    float u1 = ((h >> 8) + 1u) * (1.0f / 16777216.0f);
    float u2 = (h2 >> 8) * (1.0f / 16777216.0f);
    return sqrtf(-2.0f * __logf(u1)) * __cosf(6.28318530718f * u2);
}

__device__ inline int lbound64(const float* a, float q) {
    int lo = 0, hi = 64;
    while (lo < hi) { int mid = (lo + hi) >> 1; if (a[mid] < q) lo = mid + 1; else hi = mid; }
    return lo < 63 ? lo : 63;
}

// One thread per (c,m) column; blockIdx.y = intensity step s (0..9).
// W_2^2 of (x_pdf+dist, y_pdf+dist), supports arange(64) -> otd[s*C + c].
// X,Y are NHWC.
__global__ void __launch_bounds__(32)
intervK(const float* __restrict__ X, const float* __restrict__ Y,
        const float* __restrict__ scal, float* __restrict__ otd,
        int C, int M, int H, int W, int stage) {
    __shared__ float sm[32 * 256];   // per column: u[64] v[64] q[128]
    const int tid = threadIdx.x;
    const int col = blockIdx.x * 32 + tid;
    const int s   = blockIdx.y;
    if (col >= C * M) return;
    float* u = &sm[tid * 256];
    float* v = u + 64;
    float* q = u + 128;

    const int c  = col / M;
    const int m  = col - c * M;
    const int Wp = W >> 3;
    const int ph = m / Wp, pw = m - ph * Wp;
    const float s0 = scal[0], s1 = scal[1];
    const float mean  = scal[2] / (float)(C * H * W);
    const float inten = 0.01f * (float)s;

    float uc = 0.0f, vc = 0.0f;
    for (int j = 0; j < 64; ++j) {
        const int jr = j >> 3, jc = j & 7;
        const size_t pix = (size_t)((ph * 8 + jr) * W + (pw * 8 + jc));
        const float xv = X[pix * C + c];
        const float yv = Y[pix * C + c];
        const float xp = xv * xv / (s0 + 1e-6f);
        const float yp = yv * yv / (s1 + 1e-6f);
        const float d  = gaussn(stage, s, c, m, j) * mean * inten;
        uc += xp + d;
        vc += yp + d;
        u[j] = uc; v[j] = vc;
        q[j] = uc; q[64 + j] = vc;
    }
    for (int i = 1; i < 128; ++i) {            // insertion sort q[0..127]
        const float key = q[i];
        int k2 = i - 1;
        while (k2 >= 0 && q[k2] > key) { q[k2 + 1] = q[k2]; --k2; }
        q[k2 + 1] = key;
    }
    float cost = 0.0f, prev = 0.0f;
    for (int i = 0; i < 128; ++i) {
        const float qq = q[i];
        const float diff = (float)(lbound64(u, qq) - lbound64(v, qq));
        cost += (qq - prev) * diff * diff;
        prev = qq;
    }
    atomicAdd(&otd[s * C + c], cost);
}

// l1[c] = sum_pixels |x_pdf - y_pdf|   (NHWC)
__global__ void l1K(const float* __restrict__ X, const float* __restrict__ Y,
                    const float* __restrict__ scal, float* __restrict__ l1,
                    int C, int H, int W) {
    __shared__ float red[256];
    const int c = blockIdx.x;
    const int n = H * W;
    const float s0 = scal[0], s1 = scal[1];
    float p = 0.0f;
    for (int i = threadIdx.x; i < n; i += 256) {
        const float xv = X[(size_t)i * C + c];
        const float yv = Y[(size_t)i * C + c];
        p += fabsf(xv * xv / (s0 + 1e-6f) - yv * yv / (s1 + 1e-6f));
    }
    red[threadIdx.x] = p; __syncthreads();
    for (int st = 128; st > 0; st >>= 1) {
        if (threadIdx.x < st) red[threadIdx.x] += red[threadIdx.x + st];
        __syncthreads();
    }
    if (threadIdx.x == 0) l1[c] = red[0];
}

// Per-channel gate (intensity search) and stage total -> d_out (mean of 6).
__global__ void gateFinal(const float* __restrict__ otd, const float* __restrict__ l1,
                          const float* __restrict__ scal, int C, float* out) {
    __shared__ float red[512];
    const int t = threadIdx.x;
    const float thresh = fabsf(scal[3]) * 1e-4f;
    float p = 0.0f;
    for (int c = t; c < C; c += 512) {
        const float ot0 = otd[c];
        float best = 0.0f;
        for (int s = 0; s < 10; ++s)
            if (fabsf(ot0 - otd[s * C + c]) < thresh) best = fmaxf(best, 0.01f * (float)s);
        if (best == 0.0f) p += ot0 + l1[c];   // gate: cci == MAX_INTENSITY
    }
    red[t] = p; __syncthreads();
    for (int st = 256; st > 0; st >>= 1) {
        if (t < st) red[t] += red[t + st];
        __syncthreads();
    }
    if (t == 0) atomicAdd(out, red[0] * (1.0f / 6.0f));
}

// ---------------------------------------------------------------------------
// Host orchestration
// ---------------------------------------------------------------------------
static const int CIN_[13]  = {3, 64, 64, 128, 128, 256, 256, 256, 512, 512, 512, 512, 512};
static const int COUT_[13] = {64, 64, 128, 128, 256, 256, 256, 512, 512, 512, 512, 512, 512};

extern "C" void kernel_launch(void* const* d_in, const int* in_sizes, int n_in,
                              void* d_out, int out_size, void* d_ws, size_t ws_size,
                              hipStream_t stream) {
    (void)in_sizes; (void)n_in; (void)out_size; (void)ws_size;
    const float* x = (const float*)d_in[0];
    const float* y = (const float*)d_in[1];
    const float* W_[13];
    const float* B_[13];
    for (int i = 0; i < 13; ++i) { W_[i] = (const float*)d_in[2 + i]; B_[i] = (const float*)d_in[15 + i]; }

    char* ws = (char*)d_ws;

    // Accumulator region
    float* scal = (float*)ws;         // 4
    float* otd  = scal + 4;           // 10*512
    float* l1   = otd + 5120;         // 512
    const int ACCF = 4 + 5120 + 512;
    size_t cur = 24576;               // 24 KB, 256-aligned

    // Packed weights (channel dim padded to mult of 32 for layer 0)
    _Float16* pk[13];
    int Kc_[13];                      // KcC = padded Cin / 32
    for (int i = 0; i < 13; ++i) {
        const int Cinp = (i == 0) ? 32 : CIN_[i];
        Kc_[i] = Cinp / 32;
        pk[i] = (_Float16*)(ws + cur);
        cur += (size_t)(COUT_[i] / 16) * (9 * Kc_[i]) * 512 * sizeof(_Float16);
    }

    // Activation buffers (NHWC; max 128*128*64 elements)
    const size_t ACT = 128 * 128 * 64;
    _Float16* a16[2]; _Float16* b16[2]; float* feat[2];
    for (int img = 0; img < 2; ++img) {
        a16[img]  = (_Float16*)(ws + cur); cur += ACT * sizeof(_Float16);
        b16[img]  = (_Float16*)(ws + cur); cur += ACT * sizeof(_Float16);
        feat[img] = (float*)(ws + cur);    cur += ACT * sizeof(float);
    }

    zeroK<<<1, 32, 0, stream>>>((float*)d_out, 1);

    // Repack all weights
    for (int i = 0; i < 13; ++i) {
        const int total = (COUT_[i] / 16) * (9 * Kc_[i]) * 512;
        repackW<<<(total + 255) / 256, 256, 0, stream>>>(W_[i], pk[i], COUT_[i], CIN_[i], Kc_[i], total);
    }

    // Layer-0 NHWC padded f16 inputs + NHWC f32 copies for stage-0 distance
    padInputNHWC16<<<(128 * 128 * 32 + 255) / 256, 256, 0, stream>>>(x, a16[0], 128, 128);
    padInputNHWC16<<<(128 * 128 * 32 + 255) / 256, 256, 0, stream>>>(y, a16[1], 128, 128);
    nchwToNhwcF32<<<(3 * 128 * 128 + 255) / 256, 256, 0, stream>>>(x, feat[0], 3, 128, 128);
    nchwToNhwcF32<<<(3 * 128 * 128 + 255) / 256, 256, 0, stream>>>(y, feat[1], 3, 128, 128);

    auto runDist = [&](const float* Xf, const float* Yf, int C, int H, int Wd, int stage) {
        const int M = (H / 8) * (Wd / 8);
        const int N = C * H * Wd;
        zeroK<<<(ACCF + 255) / 256, 256, 0, stream>>>(scal, ACCF);
        int rg = (N + 255) / 256; if (rg > 1024) rg = 1024;
        reduceRaw<<<rg, 256, 0, stream>>>(Xf, Yf, N, scal);
        reducePdf<<<rg, 256, 0, stream>>>(Xf, Yf, N, scal);
        dim3 ig((C * M + 31) / 32, 10);
        intervK<<<ig, 32, 0, stream>>>(Xf, Yf, scal, otd, C, M, H, Wd, stage);
        l1K<<<C, 256, 0, stream>>>(Xf, Yf, scal, l1, C, H, Wd);
        gateFinal<<<1, 512, 0, stream>>>(otd, l1, scal, C, (float*)d_out);
    };

    auto runConv = [&](int i, const _Float16* in16, _Float16* out16, float* f32, int H) {
        const int Cinp = (i == 0) ? 32 : CIN_[i];
        dim3 g((H * H) / 64, COUT_[i] / 64);
        convWMMA<<<g, 128, 0, stream>>>(in16, pk[i], B_[i], f32, out16,
                                        H, H, Cinp, Kc_[i], COUT_[i]);
    };

    // Stage 0 distance on raw inputs (NHWC copies)
    runDist(feat[0], feat[1], 3, 128, 128, 0);

    const int stageN[5] = {2, 2, 3, 3, 3};
    const int stage0[5] = {0, 2, 4, 7, 10};
    _Float16* curB[2] = {a16[0], a16[1]};
    _Float16* altB[2] = {b16[0], b16[1]};
    int H = 128;

    for (int s = 0; s < 5; ++s) {
        if (s > 0) {
            const int Cp = COUT_[stage0[s] - 1];
            const int tot = Cp * (H / 2) * (H / 2);
            for (int img = 0; img < 2; ++img)
                l2poolK<<<(tot + 255) / 256, 256, 0, stream>>>(feat[img], curB[img], Cp, H, H);
            H >>= 1;
        }
        for (int img = 0; img < 2; ++img) {
            for (int k = 0; k < stageN[s]; ++k) {
                runConv(stage0[s] + k, curB[img], altB[img], feat[img], H);
                _Float16* t = curB[img]; curB[img] = altB[img]; altB[img] = t;
            }
        }
        runDist(feat[0], feat[1], COUT_[stage0[s] + stageN[s] - 1], H, H, s + 1);
    }
}
